// LoRALinear_71090298683825
// MI455X (gfx1250) — compile-verified, matrix operand order checked by reading, not run on hardware
//
#include <hip/hip_runtime.h>

// LoRA-merged GEMM for MI455X (gfx1250, wave32, WMMA).
//   out[T,N] = x[T,K] @ (W + s1*B1@A1 + s2*B2@A2)^T[K,N] + b
// T=32768, K=N=1024, RANK=16, s1=8/16=0.5, s2=16/16=1.0
//
// Roofline: 68.7 GFLOP GEMM; min HBM traffic 128MB x(f32) + 128MB out(f32)
// + 2MB W'(f16, L2-resident) ~ 258MB -> ~11us @ 23.3 TB/s. f16 WMMA keeps
// compute in the same ballpark -> balanced; merged weight kills the two
// extra x passes a naive 3-GEMM LoRA would need.

#define TOKENS 32768
#define K_DIM  1024
#define N_DIM  1024
#define RANK   16
#define SCALE1 0.5f
#define SCALE2 1.0f

typedef __attribute__((ext_vector_type(16))) _Float16 v16h;
typedef __attribute__((ext_vector_type(8)))  _Float16 v8h;
typedef __attribute__((ext_vector_type(4)))  _Float16 v4h;
typedef __attribute__((ext_vector_type(8)))  float    v8f;
typedef __attribute__((ext_vector_type(4)))  float    v4f;

// ---------------------------------------------------------------------------
// Kernel 1: merge LoRA adapters into the base weight, emit f16 W' into d_ws.
// One block per output row o (1024 blocks); thread t handles columns 4t..4t+3.
// B1/B2 row reads are wave-uniform -> scalar loads; A1/A2 reads are b128.
// ---------------------------------------------------------------------------
__global__ __launch_bounds__(256) void lora_merge_w(
    const float* __restrict__ W,
    const float* __restrict__ A1, const float* __restrict__ B1,
    const float* __restrict__ A2, const float* __restrict__ B2,
    _Float16* __restrict__ wh)
{
    const int o = blockIdx.x;            // output row, 0..1023
    const int i = threadIdx.x * 4;       // column start, 0..1020

    v4f acc1 = {0.f, 0.f, 0.f, 0.f};
    v4f acc2 = {0.f, 0.f, 0.f, 0.f};
#pragma unroll
    for (int r = 0; r < RANK; ++r) {
        const float b1 = B1[o * RANK + r];          // uniform -> s_load
        const float b2 = B2[o * RANK + r];
        const v4f a1 = *(const v4f*)(A1 + (size_t)r * K_DIM + i);
        const v4f a2 = *(const v4f*)(A2 + (size_t)r * K_DIM + i);
        acc1 += a1 * b1;
        acc2 += a2 * b2;
    }
    const v4f w = *(const v4f*)(W + (size_t)o * K_DIM + i);
    const v4f res = w + SCALE1 * acc1 + SCALE2 * acc2;

    v4h h;
#pragma unroll
    for (int j = 0; j < 4; ++j) h[j] = (_Float16)res[j];
    *(v4h*)(wh + (size_t)o * K_DIM + i) = h;        // b64 store
}

// ---------------------------------------------------------------------------
// Fragment builders.
// ISA 16-bit A 16x32 layout: lane l<16 -> row M=l, K = k0..k0+7 (v0..v3) and
// k0+16..k0+23 (v4..v7); lanes 16..31 use k0 = kb+8. So per lane the fragment
// is two contiguous 8-element runs -> two 16B loads (f32: four v4f + cvt).
// B (K x 16) mirrors this with N in the lane dim; since B = W'^T, a B "column"
// is a contiguous W' row -> same two-contiguous-chunk pattern on f16 data.
// ---------------------------------------------------------------------------
__device__ __forceinline__ v16h load_a_frag_f32(const float* __restrict__ p)
{
    const v4f f0 = *(const v4f*)(p);
    const v4f f1 = *(const v4f*)(p + 4);
    const v4f f2 = *(const v4f*)(p + 16);
    const v4f f3 = *(const v4f*)(p + 20);
    v16h a;
#pragma unroll
    for (int j = 0; j < 4; ++j) {
        a[j]      = (_Float16)f0[j];   // K = k0 .. k0+3
        a[j + 4]  = (_Float16)f1[j];   // K = k0+4 .. k0+7
        a[j + 8]  = (_Float16)f2[j];   // K = k0+16 .. k0+19
        a[j + 12] = (_Float16)f3[j];   // K = k0+20 .. k0+23
    }
    return a;
}

__device__ __forceinline__ v16h load_b_frag_f16(const _Float16* __restrict__ p)
{
    const v8h lo = *(const v8h*)(p);        // K = k0 .. k0+7   (b128)
    const v8h hi = *(const v8h*)(p + 16);   // K = k0+16..k0+23 (b128)
    return __builtin_shufflevector(lo, hi,
        0, 1, 2, 3, 4, 5, 6, 7, 8, 9, 10, 11, 12, 13, 14, 15);
}

// ---------------------------------------------------------------------------
// Kernel 2: tiled WMMA GEMM. Block = 256 threads = 8 waves -> 128x128 C tile.
// Wave (wr,wc) owns a 32x64 subtile = 2x4 accumulators of v_wmma_f32_16x16x32_f16.
// ---------------------------------------------------------------------------
__global__ __launch_bounds__(256) void lora_gemm_f16(
    const float* __restrict__ x,
    const _Float16* __restrict__ wh,
    const float* __restrict__ bias,
    float* __restrict__ out)
{
    const int lane = threadIdx.x & 31;
    const int wave = threadIdx.x >> 5;
    const int half = lane >> 4;     // selects K-offset 0 or 8 within fragment
    const int l    = lane & 15;     // row (A) / col (B,C) within 16-wide tile

    const int wr = wave & 3;        // 4 row groups of waves
    const int wc = wave >> 2;       // 2 col groups of waves

    const int row0 = blockIdx.y * 128 + wr * 32;   // wave rows [row0, row0+32)
    const int col0 = blockIdx.x * 128 + wc * 64;   // wave cols [col0, col0+64)

    const float* xr0 = x + (size_t)(row0 + l)      * K_DIM;
    const float* xr1 = x + (size_t)(row0 + 16 + l) * K_DIM;
    const _Float16* wp0 = wh + (size_t)(col0 +  0 + l) * K_DIM;
    const _Float16* wp1 = wh + (size_t)(col0 + 16 + l) * K_DIM;
    const _Float16* wp2 = wh + (size_t)(col0 + 32 + l) * K_DIM;
    const _Float16* wp3 = wh + (size_t)(col0 + 48 + l) * K_DIM;

    v8f acc[2][4] = {};   // 64 VGPRs of f32 C

    for (int kb = 0; kb < K_DIM; kb += 32) {
        const int k0 = kb + half * 8;

        // Unconditional speculative prefetch of the next K-slice (locality 3
        // -> WGP scope: pull into all cache levels). OOB translation failures
        // on the final iteration are silently dropped per the ISA; no branch.
        __builtin_prefetch((const void*)(xr0 + kb + 32), 0, 3);
        __builtin_prefetch((const void*)(xr1 + kb + 32), 0, 3);
        __builtin_prefetch((const void*)(wp0 + kb + 64), 0, 3);
        __builtin_prefetch((const void*)(wp1 + kb + 64), 0, 3);
        __builtin_prefetch((const void*)(wp2 + kb + 64), 0, 3);
        __builtin_prefetch((const void*)(wp3 + kb + 64), 0, 3);

        v16h afrag[2];
        afrag[0] = load_a_frag_f32(xr0 + k0);
        afrag[1] = load_a_frag_f32(xr1 + k0);

        v16h bfrag[4];
        bfrag[0] = load_b_frag_f16(wp0 + k0);
        bfrag[1] = load_b_frag_f16(wp1 + k0);
        bfrag[2] = load_b_frag_f16(wp2 + k0);
        bfrag[3] = load_b_frag_f16(wp3 + k0);

#pragma unroll
        for (int r = 0; r < 2; ++r)
#pragma unroll
            for (int c = 0; c < 4; ++c)
                acc[r][c] = __builtin_amdgcn_wmma_f32_16x16x32_f16(
                    /*neg_a=*/false, afrag[r],
                    /*neg_b=*/false, bfrag[c],
                    /*c_mod=*/(short)0, acc[r][c],
                    /*reuse_a=*/false, /*reuse_b=*/false);
    }

    // Epilogue: C layout -> lane holds column n = col0+16c+l, rows
    // m = row0 + 16r + half*8 + j (j = VGPR index 0..7). Add bias, store f32.
#pragma unroll
    for (int c = 0; c < 4; ++c) {
        const int n = col0 + 16 * c + l;
        const float bv = bias[n];
#pragma unroll
        for (int r = 0; r < 2; ++r) {
            const int m0 = row0 + 16 * r + half * 8;
#pragma unroll
            for (int j = 0; j < 8; ++j)
                out[(size_t)(m0 + j) * N_DIM + n] = acc[r][c][j] + bv;
        }
    }
}

// ---------------------------------------------------------------------------
// Host launcher. d_ws usage: 1024*1024 f16 merged weight = 2 MB.
// ---------------------------------------------------------------------------
extern "C" void kernel_launch(void* const* d_in, const int* in_sizes, int n_in,
                              void* d_out, int out_size, void* d_ws, size_t ws_size,
                              hipStream_t stream) {
    const float* x  = (const float*)d_in[0];
    const float* W  = (const float*)d_in[1];
    const float* b  = (const float*)d_in[2];
    const float* A1 = (const float*)d_in[3];
    const float* B1 = (const float*)d_in[4];
    const float* A2 = (const float*)d_in[5];
    const float* B2 = (const float*)d_in[6];
    float* out = (float*)d_out;

    _Float16* wh = (_Float16*)d_ws;   // 2 MB merged f16 weight (L2-resident)

    // Fold LoRA adapters into the base weight (trivial cost, kills 2 GEMMs).
    lora_merge_w<<<dim3(N_DIM), dim3(256), 0, stream>>>(W, A1, B1, A2, B2, wh);

    // Single WMMA GEMM: grid = (1024/128 cols, 32768/128 rows).
    lora_gemm_f16<<<dim3(N_DIM / 128, TOKENS / 128), dim3(256), 0, stream>>>(
        x, wh, b, out);
}
